// MultiDomainPLEFENDModel_38147899523621
// MI455X (gfx1250) — matrix operand order repaired and checked by reading, not compiled, not run on hardware
//
#include <hip/hip_runtime.h>
#include <hip/hip_bf16.h>
#include <math.h>

// ---------------------------------------------------------------------------
// MultiDomainPLEFEND forward for MI455X (gfx1250, wave32, WMMA).
// Dominant work: two fused GEMM+token-maxpool ops via v_wmma_f32_16x16x32_bf16,
// with register-resident B panels (192 VGPRs/lane) and LDS-staged A tiles
// (async global->LDS when available). Per-expert fusion MLP via WMMA GEMM.
// ---------------------------------------------------------------------------

#define BB   64
#define TT   197
#define IT   198
#define TPAD 208      // tokens padded to multiple of 16
#define DD   768
#define EE   18
#define HH   320
#define PP   12
#define NEXP (EE*HH)  // 5760

typedef __bf16 bf16_t;
typedef __attribute__((ext_vector_type(16))) __bf16 v16bf;
typedef __attribute__((ext_vector_type(8)))  __bf16 v8bf;
typedef __attribute__((ext_vector_type(8)))  float  v8f;

typedef __attribute__((__vector_size__(4 * sizeof(int)))) int i32x4;
typedef __attribute__((address_space(1))) i32x4 gi32x4;   // global
typedef __attribute__((address_space(3))) i32x4 li32x4;   // LDS

#ifndef __has_builtin
#define __has_builtin(x) 0
#endif
#if __has_builtin(__builtin_amdgcn_global_load_async_to_lds_b128) && \
    __has_builtin(__builtin_amdgcn_s_wait_asynccnt)
#define ASYNC_LDS 1
#else
#define ASYNC_LDS 0
#endif

// ---------------- helpers ----------------
__device__ inline bf16_t f2bf(float f) {
  union { float f; unsigned u; } v; v.f = f;
  unsigned r = v.u + 0x7FFFu + ((v.u >> 16) & 1u);   // round-to-nearest-even
  unsigned short h = (unsigned short)(r >> 16);
  return __builtin_bit_cast(bf16_t, h);
}

__device__ inline float act_apply(float x, int act) {
  switch (act) {
    case 1: return fmaxf(x, 0.0f);                                   // relu
    case 2: return x / (1.0f + __expf(-x));                          // silu
    case 3: { float x3 = x * x * x;                                  // gelu(tanh)
              return 0.5f * x * (1.0f + tanhf(0.7978845608f * (x + 0.044715f * x3))); }
    default: return x;
  }
}

// A-fragment (16x32 bf16): lane L -> row = L%16; K = k0 + off + {0..7, 16..23}, off = (L<16)?0:8
__device__ inline v16bf load_a_frag(const bf16_t* __restrict__ rowptr_koff, int k0) {
  v8bf lo = *(const v8bf*)(rowptr_koff + k0);
  v8bf hi = *(const v8bf*)(rowptr_koff + k0 + 16);
  v16bf r;
#pragma unroll
  for (int i = 0; i < 8; ++i) { r[i] = lo[i]; r[i + 8] = hi[i]; }
  return r;
}

// B-fragment (32x16 bf16): lane L -> col = L%16; K = k0 + koff + {0..7, 16..23}, koff = (L<16)?0:8
__device__ inline v16bf load_b_frag(const bf16_t* __restrict__ colptr_koff, int k0) {
  v8bf lo = *(const v8bf*)(colptr_koff + k0);
  v8bf hi = *(const v8bf*)(colptr_koff + k0 + 16);
  v16bf r;
#pragma unroll
  for (int i = 0; i < 8; ++i) { r[i] = lo[i]; r[i + 8] = hi[i]; }
  return r;
}

// cooperative 24 KB tile copy, 256 threads x 6 x 16B; async global->LDS on CDNA5
__device__ inline void copy_tile_to_lds(bf16_t* __restrict__ smem,
                                        const bf16_t* __restrict__ gsrc, int tid) {
#pragma unroll
  for (int j = 0; j < 6; ++j) {
    const int byte = (j * 256 + tid) * 16;
#if ASYNC_LDS
    __builtin_amdgcn_global_load_async_to_lds_b128(
        (gi32x4*)((char*)gsrc + byte),
        (li32x4*)((char*)smem + byte), 0, 0);
#else
    *(uint4*)((char*)smem + byte) = *(const uint4*)((const char*)gsrc + byte);
#endif
  }
#if ASYNC_LDS
  __builtin_amdgcn_s_wait_asynccnt(0);
#endif
}

// ---------------- conversion kernels ----------------
// f32 [B,T,D] -> bf16 [B,TPAD,D], zero padded rows
__global__ void cvt_seq_kernel(const float* __restrict__ in, bf16_t* __restrict__ out, int T) {
  long idx = (long)blockIdx.x * blockDim.x + threadIdx.x;
  long total = (long)BB * TPAD * DD;
  if (idx >= total) return;
  int d = (int)(idx % DD);
  int t = (int)((idx / DD) % TPAD);
  int b = (int)(idx / ((long)DD * TPAD));
  out[idx] = (t < T) ? f2bf(in[((long)b * T + t) * DD + d]) : f2bf(0.0f);
}

// f32 [Z][K,N] -> bf16 [Z][N,K] (weight transpose per batch/expert)
__global__ void cvt_wT_kernel(const float* __restrict__ in, bf16_t* __restrict__ out, int K, int N) {
  int z = blockIdx.z;
  long idx = (long)blockIdx.x * blockDim.x + threadIdx.x;
  long total = (long)N * K;
  if (idx >= total) return;
  int k = (int)(idx % K);
  int n = (int)(idx / K);
  out[(long)z * total + idx] = f2bf(in[(long)z * total + (long)k * N + n]);
}

// flat f32 -> bf16
__global__ void cvt_flat_kernel(const float* __restrict__ in, bf16_t* __restrict__ out, long n) {
  long idx = (long)blockIdx.x * blockDim.x + threadIdx.x;
  if (idx < n) out[idx] = f2bf(in[idx]);
}

// ---------------- dominant kernel: GEMM + fused token max-pool ----------------
// out[b, n] = max_t ( sum_k A[b,t,k] * W[n,k] ) + bias[n]
// Workgroup = 8 waves: 128-column stripe x 8 batches.
// B panel (16 cols x 768 K = 24 KB) lives in VGPRs per wave, loaded once.
// A tile (16 x 768 = 24 KB) staged in LDS per (b, t0), shared by all 8 waves.
__global__ void __launch_bounds__(256)
wmma_maxpool_kernel(const bf16_t* __restrict__ A,   // [B, TPAD, 768]
                    const bf16_t* __restrict__ W,   // [N, 768]
                    const float*  __restrict__ bias,// [N]
                    float*        __restrict__ out, // [B, N]
                    int T, int N) {
  constexpr int K = DD;
  const int tid   = threadIdx.x;
  const int lane  = tid & 31;
  const int wave  = tid >> 5;
  const int n     = blockIdx.x * 128 + wave * 16 + (lane & 15);
  const int b0    = blockIdx.y * 8;

  __shared__ __align__(16) bf16_t smemA[16 * DD];   // 24 KB

  // preload this wave's B panel into registers (24 frags = 192 VGPRs/lane)
  const int koff = (lane < 16) ? 0 : 8;
  const bf16_t* Bn = W + (long)n * K + koff;
  v16bf breg[K / 32];
#pragma unroll
  for (int kk = 0; kk < K / 32; ++kk) breg[kk] = load_b_frag(Bn, kk * 32);

  const int aoff  = (lane < 16) ? 0 : 8;
  const int mhalf = (lane < 16) ? 0 : 8;
  const float bn  = bias[n];

  for (int bi = 0; bi < 8; ++bi) {
    const int b = b0 + bi;
    const bf16_t* Ab = A + (long)b * TPAD * K;
    float runm[8];
#pragma unroll
    for (int r = 0; r < 8; ++r) runm[r] = -3.0e38f;

    for (int t0 = 0; t0 < TPAD; t0 += 16) {
      __syncthreads();                                  // previous tile fully consumed
      copy_tile_to_lds(smemA, Ab + (long)t0 * K, tid);
      __syncthreads();                                  // tile visible to all waves

      const bf16_t* Ar = smemA + (lane & 15) * K + aoff;
      v8f c = {};
#pragma unroll
      for (int kk = 0; kk < K / 32; ++kk) {
        v16bf a = load_a_frag(Ar, kk * 32);             // ds_load_b128 x2
        c = __builtin_amdgcn_wmma_f32_16x16x32_bf16(false, a, false, breg[kk],
                                                    (short)0, c, false, false);
      }
      const int mb = t0 + mhalf;
#pragma unroll
      for (int r = 0; r < 8; ++r) {
        float v = (mb + r < T) ? c[r] : -3.0e38f;
        runm[r] = fmaxf(runm[r], v);
      }
    }
    float m = runm[0];
#pragma unroll
    for (int r = 1; r < 8; ++r) m = fmaxf(m, runm[r]);
    m = fmaxf(m, __shfl_xor(m, 16, 32));                // combine M-halves across lane^16
    if (lane < 16) out[(long)b * N + n] = m + bn;
  }
}

// ---------------- generic WMMA GEMM: out = act(A @ W^T + bias) ----------------
__global__ void wmma_gemm_kernel(const bf16_t* __restrict__ A, long a_zstride,   // [M,K]
                                 const bf16_t* __restrict__ W, long w_zstride,   // [N,K]
                                 const float*  __restrict__ bias, long b_zstride,// [N]
                                 float*        __restrict__ out, long o_zstride, // [M,N]
                                 int M, int N, int K, int act) {
  const int lane = threadIdx.x & 31;
  const int wave = threadIdx.x >> 5;
  const int tiles_n = N >> 4;
  const int tile = blockIdx.x * (blockDim.x >> 5) + wave;
  if (tile >= (M >> 4) * tiles_n) return;
  const int mtile = tile / tiles_n, ntile = tile % tiles_n;
  const int z = blockIdx.z;
  A    += (long)z * a_zstride;
  W    += (long)z * w_zstride;
  bias += (long)z * b_zstride;
  out  += (long)z * o_zstride;

  const int n    = ntile * 16 + (lane & 15);
  const int aoff = (lane < 16) ? 0 : 8;
  const int koff = (lane < 16) ? 0 : 8;
  const bf16_t* Ar = A + (long)(mtile * 16 + (lane & 15)) * K + aoff;
  const bf16_t* Bn = W + (long)n * K + koff;

  v8f c = {};
  for (int k0 = 0; k0 < K; k0 += 32) {
    v16bf a = load_a_frag(Ar, k0);
    v16bf w = load_b_frag(Bn, k0);
    c = __builtin_amdgcn_wmma_f32_16x16x32_bf16(false, a, false, w, (short)0, c, false, false);
  }
  const int m0 = mtile * 16 + ((lane < 16) ? 0 : 8);
  const float bn = bias[n];
#pragma unroll
  for (int r = 0; r < 8; ++r)
    out[(long)(m0 + r) * N + n] = act_apply(c[r] + bn, act);
}

// ---------------- token attention pooling ----------------
__global__ void attn_pool_kernel(const float* __restrict__ seq, const float* __restrict__ mask,
                                 const float* __restrict__ w, const float* __restrict__ bias,
                                 float* __restrict__ out, int T) {
  const int b = blockIdx.x, tid = threadIdx.x;
  __shared__ float prob[256];
  __shared__ float red[256];
  float l = -3.0e38f;
  if (tid < T) {
    const float* row = seq + ((long)b * T + tid) * DD;
    float acc = bias[0];
    for (int d = 0; d < DD; ++d) acc += row[d] * w[d];
    l = mask ? ((mask[(long)b * T + tid] > 0.0f) ? acc : -1.0e9f) : acc;
  }
  red[tid] = l; __syncthreads();
  for (int s = 128; s > 0; s >>= 1) { if (tid < s) red[tid] = fmaxf(red[tid], red[tid + s]); __syncthreads(); }
  const float mx = red[0]; __syncthreads();
  float e = (tid < T) ? __expf(l - mx) : 0.0f;
  prob[tid] = e; red[tid] = e; __syncthreads();
  for (int s = 128; s > 0; s >>= 1) { if (tid < s) red[tid] += red[tid + s]; __syncthreads(); }
  const float inv = 1.0f / red[0]; __syncthreads();
  for (int d = tid; d < DD; d += blockDim.x) {
    float acc = 0.0f;
    for (int t = 0; t < T; ++t) acc += prob[t] * seq[((long)b * T + t) * DD + d];
    out[(long)b * DD + d] = acc * inv;
  }
}

// ---------------- small dense layers (f32, one thread per output) ----------------
__global__ void linear_kernel(const float* __restrict__ in, const float* __restrict__ W,
                              const float* __restrict__ bias, float* __restrict__ out,
                              int K, int N, int act) {
  const int j = blockIdx.x * blockDim.x + threadIdx.x;
  const int b = blockIdx.y;
  if (j >= N) return;
  const float* x = in + (long)b * K;
  float acc = bias ? bias[j] : 0.0f;
  for (int k = 0; k < K; ++k) acc += x[k] * W[(long)k * N + j];
  out[(long)b * N + j] = act_apply(acc, act);
}

__global__ void softmax_rows_kernel(float* __restrict__ x, int n) {
  const int b = blockIdx.x * blockDim.x + threadIdx.x;
  if (b >= BB) return;
  float* row = x + (long)b * n;
  float mx = -3.0e38f;
  for (int i = 0; i < n; ++i) mx = fmaxf(mx, row[i]);
  float s = 0.0f;
  for (int i = 0; i < n; ++i) { float e = __expf(row[i] - mx); row[i] = e; s += e; }
  const float inv = 1.0f / s;
  for (int i = 0; i < n; ++i) row[i] *= inv;
}

// out[b,k] = sum_e g[b,e] * ex[e*es + b*bs + k]
__global__ void expert_sum_kernel(const float* __restrict__ g, const float* __restrict__ ex,
                                  float* __restrict__ out, long es, long bs) {
  const int k = blockIdx.x * blockDim.x + threadIdx.x;
  const int b = blockIdx.y;
  if (k >= HH) return;
  float acc = 0.0f;
  for (int e = 0; e < EE; ++e) acc += g[(long)b * EE + e] * ex[(long)e * es + (long)b * bs + k];
  out[(long)b * HH + k] = acc;
}

// out[b,:] = softmax(l2[b,:])[0] * x[b,:]
__global__ void scale01_kernel(const float* __restrict__ l2, const float* __restrict__ x,
                               float* __restrict__ out) {
  const int k = blockIdx.x * blockDim.x + threadIdx.x;
  const int b = blockIdx.y;
  if (k >= HH) return;
  const float a = l2[(long)b * 2], c = l2[(long)b * 2 + 1];
  const float mx = fmaxf(a, c);
  const float e0 = __expf(a - mx), e1 = __expf(c - mx);
  out[(long)b * HH + k] = (e0 / (e0 + e1)) * x[(long)b * HH + k];
}

__global__ void concat3_kernel(const float* __restrict__ a, int wa,
                               const float* __restrict__ b_, int wb,
                               const float* __restrict__ c, int wc,
                               float* __restrict__ out) {
  const int W = wa + wb + wc;
  const int j = blockIdx.x * blockDim.x + threadIdx.x;
  const int b = blockIdx.y;
  if (j >= W) return;
  float v;
  if (j < wa)            v = a[(long)b * wa + j];
  else if (j < wa + wb)  v = b_[(long)b * wb + (j - wa)];
  else                   v = c[(long)b * wc + (j - wa - wb)];
  out[(long)b * W + j] = v;
}

// prototype similarities + squared cost  C[b,i,j] = (tf.tp_i - if.ip_j)^2
__global__ void proto_cost_kernel(const float* __restrict__ tf, const float* __restrict__ imf,
                                  const float* __restrict__ tp, const float* __restrict__ ip,
                                  float* __restrict__ C) {
  const int b = blockIdx.x, tid = threadIdx.x;
  __shared__ float st[PP], si[PP];
  if (tid < PP) {
    float acc = 0.0f;
    for (int k = 0; k < HH; ++k) acc += tf[(long)b * HH + k] * tp[(long)tid * HH + k];
    st[tid] = acc;
  } else if (tid < 2 * PP) {
    const int p = tid - PP;
    float acc = 0.0f;
    for (int k = 0; k < HH; ++k) acc += imf[(long)b * HH + k] * ip[(long)p * HH + k];
    si[p] = acc;
  }
  __syncthreads();
  if (tid < PP * PP) {
    const int i = tid / PP, j = tid % PP;
    const float d = st[i] - si[j];
    C[(long)b * PP * PP + tid] = d * d;
  }
}

__global__ void sinkhorn_kernel(const float* __restrict__ C, float* __restrict__ Tm,
                                int iters, float eps) {
  const int b = blockIdx.x, tid = threadIdx.x;   // blockDim = 32
  __shared__ float Cn[PP * PP], u[PP], v[PP], mx;
  if (tid == 0) {
    float m = -3.0e38f;
    for (int i = 0; i < PP * PP; ++i) m = fmaxf(m, C[(long)b * PP * PP + i]);
    mx = m;
  }
  __syncthreads();
  const float inv = 1.0f / (mx + 1e-8f);
  for (int i = tid; i < PP * PP; i += 32) Cn[i] = C[(long)b * PP * PP + i] * inv;
  if (tid < PP) { u[tid] = 0.0f; v[tid] = 0.0f; }
  __syncthreads();
  const float logmu = -logf((float)PP);
  const float ieps = 1.0f / eps;
  for (int it = 0; it < iters; ++it) {
    float un = 0.0f;
    if (tid < PP) {
      const int i = tid;
      float m = -3.0e38f;
      for (int j = 0; j < PP; ++j) m = fmaxf(m, -Cn[i * PP + j] * ieps + v[j]);
      float s = 0.0f;
      for (int j = 0; j < PP; ++j) s += __expf(-Cn[i * PP + j] * ieps + v[j] - m);
      un = eps * (logmu - (m + logf(s))) + v[i];
    }
    __syncthreads();
    if (tid < PP) u[tid] = un;
    __syncthreads();
    if (tid < PP) {
      const int j = tid;
      float m = -3.0e38f;
      for (int i = 0; i < PP; ++i) m = fmaxf(m, -Cn[i * PP + j] * ieps + u[i]);
      float s = 0.0f;
      for (int i = 0; i < PP; ++i) s += __expf(-Cn[i * PP + j] * ieps + u[i] - m);
      v[j] = eps * (logmu - (m + logf(s))) + u[j];
    }
    __syncthreads();
  }
  for (int t = tid; t < PP * PP; t += 32) {
    const int i = t / PP, j = t % PP;
    Tm[(long)b * PP * PP + t] = __expf((-Cn[t] + u[i] + v[j]) * ieps);
  }
}

// ---------------------------------------------------------------------------
// input indices (setup_inputs order; params sorted by name)
enum {
  IN_TEXT = 0, IN_MASK, IN_IMG, IN_CLIP_I, IN_CLIP_T,
  P_AI_B1, P_AI_B2, P_AI_W1, P_AI_W2,
  P_AM_B1, P_AM_B2, P_AM_W1, P_AM_W2,
  P_AT_B1, P_AT_B2, P_AT_W1, P_AT_W2,
  P_CF_B1, P_CF_B2, P_CF_W1, P_CF_W2,
  P_CI_B1, P_CI_B2, P_CI_W1, P_CI_W2,
  P_CM_B1, P_CM_B2, P_CM_W1, P_CM_W2,
  P_CT_B1, P_CT_B2, P_CT_W1, P_CT_W2,
  P_DM_B1, P_DM_B2, P_DM_W1, P_DM_W2,
  P_FD_B1, P_FD_B2, P_FD_W1, P_FD_W2,
  P_FE_B1, P_FE_B2, P_FE_W1, P_FE_W2,
  P_FG_B1, P_FG_B2, P_FG_W1, P_FG_W2,
  P_IA_B, P_IA_W, P_IE_B, P_IE_W,
  P_IG_B1, P_IG_B2, P_IG_W1, P_IG_W2,
  P_IP,
  P_MF_B1, P_MF_B2, P_MF_W1, P_MF_W2,
  P_TA_B, P_TA_W, P_TE_B, P_TE_W,
  P_TG_B1, P_TG_B2, P_TG_W1, P_TG_W2,
  P_TP
};

extern "C" void kernel_launch(void* const* d_in, const int* in_sizes, int n_in,
                              void* d_out, int out_size, void* d_ws, size_t ws_size,
                              hipStream_t stream) {
  (void)in_sizes; (void)n_in; (void)out_size; (void)ws_size;
  auto F = [&](int i) { return (const float*)d_in[i]; };

  // ---- workspace carve-out ----
  char* ws = (char*)d_ws;
  size_t off = 0;
  auto alloc = [&](size_t bytes) -> void* {
    void* p = ws + off;
    off = (off + bytes + 255) & ~(size_t)255;
    return p;
  };
  bf16_t* abf_t  = (bf16_t*)alloc((size_t)BB * TPAD * DD * 2);
  bf16_t* abf_i  = (bf16_t*)alloc((size_t)BB * TPAD * DD * 2);
  bf16_t* wt_bf  = (bf16_t*)alloc((size_t)NEXP * DD * 2);
  bf16_t* wi_bf  = (bf16_t*)alloc((size_t)NEXP * DD * 2);
  bf16_t* few1   = (bf16_t*)alloc((size_t)EE * HH * HH * 2);
  bf16_t* few2   = (bf16_t*)alloc((size_t)EE * HH * HH * 2);
  float* te_out  = (float*)alloc((size_t)BB * NEXP * 4);
  float* ie_out  = (float*)alloc((size_t)BB * NEXP * 4);
  float* text_atn  = (float*)alloc((size_t)BB * DD * 4);
  float* image_atn = (float*)alloc((size_t)BB * DD * 4);
  float* clip_cat  = (float*)alloc((size_t)BB * 1024 * 4);
  float* cf_h      = (float*)alloc((size_t)BB * 348 * 4);
  float* clip_fuse = (float*)alloc((size_t)BB * HH * 4);
  float* tg_h = (float*)alloc((size_t)BB * DD * 4);
  float* tg   = (float*)alloc((size_t)BB * EE * 4);
  float* ig_h = (float*)alloc((size_t)BB * DD * 4);
  float* ig   = (float*)alloc((size_t)BB * EE * 4);
  float* text_sum  = (float*)alloc((size_t)BB * HH * 4);
  float* image_sum = (float*)alloc((size_t)BB * HH * 4);
  float* hat = (float*)alloc((size_t)BB * 174 * 4);
  float* hai = (float*)alloc((size_t)BB * 174 * 4);
  float* ham = (float*)alloc((size_t)BB * 174 * 4);
  float* lt2 = (float*)alloc((size_t)BB * 2 * 4);
  float* li2 = (float*)alloc((size_t)BB * 2 * 4);
  float* lm2 = (float*)alloc((size_t)BB * 2 * 4);
  float* text_final  = (float*)alloc((size_t)BB * HH * 4);
  float* image_final = (float*)alloc((size_t)BB * HH * 4);
  float* fusion_cat  = (float*)alloc((size_t)BB * 960 * 4);
  float* mf_h        = (float*)alloc((size_t)BB * 348 * 4);
  float* fusion      = (float*)alloc((size_t)BB * HH * 4);
  float* fg_h = (float*)alloc((size_t)BB * 160 * 4);
  float* fg   = (float*)alloc((size_t)BB * EE * 4);
  bf16_t* fusion_bf = (bf16_t*)alloc((size_t)BB * HH * 2);
  float*  fh        = (float*)alloc((size_t)EE * BB * HH * 4);
  bf16_t* fh_bf     = (bf16_t*)alloc((size_t)EE * BB * HH * 2);
  float*  fe_out    = (float*)alloc((size_t)EE * BB * HH * 4);
  float* fusion_sum   = (float*)alloc((size_t)BB * HH * 4);
  float* fusion_final = (float*)alloc((size_t)BB * HH * 4);
  float* Cmat = (float*)alloc((size_t)BB * PP * PP * 4);
  float* Tmat = (float*)alloc((size_t)BB * PP * PP * 4);
  float* dm_h = (float*)alloc((size_t)BB * 128 * 4);
  float* diss = (float*)alloc((size_t)BB * HH * 4);
  float* final_cat = (float*)alloc((size_t)BB * 640 * 4);
  float* fd_h      = (float*)alloc((size_t)BB * HH * 4);
  float* final_    = (float*)alloc((size_t)BB * HH * 4);
  float* ct_h = (float*)alloc((size_t)BB * 348 * 4);
  float* ci_h = (float*)alloc((size_t)BB * 348 * 4);
  float* cm_h = (float*)alloc((size_t)BB * 348 * 4);
  float* lt = (float*)alloc((size_t)BB * 4);
  float* li = (float*)alloc((size_t)BB * 4);
  float* lm = (float*)alloc((size_t)BB * 4);

  // ---- stage bf16 operands ----
  {
    long n = (long)BB * TPAD * DD;
    cvt_seq_kernel<<<dim3((n + 255) / 256), 256, 0, stream>>>(F(IN_TEXT), abf_t, TT);
    cvt_seq_kernel<<<dim3((n + 255) / 256), 256, 0, stream>>>(F(IN_IMG), abf_i, IT);
  }
  cvt_wT_kernel<<<dim3((DD * HH + 255) / 256, 1, EE), 256, 0, stream>>>(F(P_TE_W), wt_bf, DD, HH);
  cvt_wT_kernel<<<dim3((DD * HH + 255) / 256, 1, EE), 256, 0, stream>>>(F(P_IE_W), wi_bf, DD, HH);
  cvt_wT_kernel<<<dim3((HH * HH + 255) / 256, 1, EE), 256, 0, stream>>>(F(P_FE_W1), few1, HH, HH);
  cvt_wT_kernel<<<dim3((HH * HH + 255) / 256, 1, EE), 256, 0, stream>>>(F(P_FE_W2), few2, HH, HH);

  // ---- dominant: expert conv + token max-pool (WMMA, LDS-tiled, reg-resident B) ----
  wmma_maxpool_kernel<<<dim3(NEXP / 128, BB / 8), 256, 0, stream>>>(abf_t, wt_bf, F(P_TE_B), te_out, TT, NEXP);
  wmma_maxpool_kernel<<<dim3(NEXP / 128, BB / 8), 256, 0, stream>>>(abf_i, wi_bf, F(P_IE_B), ie_out, IT, NEXP);

  // ---- token-attention pooled features ----
  attn_pool_kernel<<<dim3(BB), 256, 0, stream>>>(F(IN_TEXT), F(IN_MASK), F(P_TA_W), F(P_TA_B), text_atn, TT);
  attn_pool_kernel<<<dim3(BB), 256, 0, stream>>>(F(IN_IMG), nullptr, F(P_IA_W), F(P_IA_B), image_atn, IT);

  // ---- clip fuse ----
  concat3_kernel<<<dim3((1024 + 127) / 128, BB), 128, 0, stream>>>(F(IN_CLIP_I), 512, F(IN_CLIP_T), 512, nullptr, 0, clip_cat);
  linear_kernel<<<dim3((348 + 127) / 128, BB), 128, 0, stream>>>(clip_cat, F(P_CF_W1), F(P_CF_B1), cf_h, 1024, 348, 1);
  linear_kernel<<<dim3((HH + 127) / 128, BB), 128, 0, stream>>>(cf_h, F(P_CF_W2), F(P_CF_B2), clip_fuse, 348, HH, 0);

  // ---- gates ----
  linear_kernel<<<dim3((DD + 127) / 128, BB), 128, 0, stream>>>(text_atn, F(P_TG_W1), F(P_TG_B1), tg_h, DD, DD, 2);
  linear_kernel<<<dim3((EE + 127) / 128, BB), 128, 0, stream>>>(tg_h, F(P_TG_W2), F(P_TG_B2), tg, DD, EE, 0);
  softmax_rows_kernel<<<dim3(1), 64, 0, stream>>>(tg, EE);
  linear_kernel<<<dim3((DD + 127) / 128, BB), 128, 0, stream>>>(image_atn, F(P_IG_W1), F(P_IG_B1), ig_h, DD, DD, 2);
  linear_kernel<<<dim3((EE + 127) / 128, BB), 128, 0, stream>>>(ig_h, F(P_IG_W2), F(P_IG_B2), ig, DD, EE, 0);
  softmax_rows_kernel<<<dim3(1), 64, 0, stream>>>(ig, EE);

  // ---- gated expert sums + per-modality attention ----
  expert_sum_kernel<<<dim3((HH + 127) / 128, BB), 128, 0, stream>>>(tg, te_out, text_sum, HH, NEXP);
  expert_sum_kernel<<<dim3((HH + 127) / 128, BB), 128, 0, stream>>>(ig, ie_out, image_sum, HH, NEXP);
  linear_kernel<<<dim3((174 + 127) / 128, BB), 128, 0, stream>>>(text_sum, F(P_AT_W1), F(P_AT_B1), hat, HH, 174, 1);
  linear_kernel<<<dim3(1, BB), 128, 0, stream>>>(hat, F(P_AT_W2), F(P_AT_B2), lt2, 174, 2, 0);
  scale01_kernel<<<dim3((HH + 127) / 128, BB), 128, 0, stream>>>(lt2, text_sum, text_final);
  linear_kernel<<<dim3((174 + 127) / 128, BB), 128, 0, stream>>>(image_sum, F(P_AI_W1), F(P_AI_B1), hai, HH, 174, 1);
  linear_kernel<<<dim3(1, BB), 128, 0, stream>>>(hai, F(P_AI_W2), F(P_AI_B2), li2, 174, 2, 0);
  scale01_kernel<<<dim3((HH + 127) / 128, BB), 128, 0, stream>>>(li2, image_sum, image_final);

  // ---- multimodal fusion ----
  concat3_kernel<<<dim3((960 + 127) / 128, BB), 128, 0, stream>>>(text_final, HH, image_final, HH, clip_fuse, HH, fusion_cat);
  linear_kernel<<<dim3((348 + 127) / 128, BB), 128, 0, stream>>>(fusion_cat, F(P_MF_W1), F(P_MF_B1), mf_h, 960, 348, 1);
  linear_kernel<<<dim3((HH + 127) / 128, BB), 128, 0, stream>>>(mf_h, F(P_MF_W2), F(P_MF_B2), fusion, 348, HH, 0);

  // ---- fusion gate + per-expert fusion MLP (WMMA batched GEMMs) ----
  linear_kernel<<<dim3((160 + 127) / 128, BB), 128, 0, stream>>>(fusion, F(P_FG_W1), F(P_FG_B1), fg_h, HH, 160, 2);
  linear_kernel<<<dim3((EE + 127) / 128, BB), 128, 0, stream>>>(fg_h, F(P_FG_W2), F(P_FG_B2), fg, 160, EE, 0);
  softmax_rows_kernel<<<dim3(1), 64, 0, stream>>>(fg, EE);
  cvt_flat_kernel<<<dim3((BB * HH + 255) / 256), 256, 0, stream>>>(fusion, fusion_bf, (long)BB * HH);
  wmma_gemm_kernel<<<dim3((BB / 16) * (HH / 16) / 4, 1, EE), 128, 0, stream>>>(
      fusion_bf, 0, few1, (long)HH * HH, F(P_FE_B1), HH, fh, (long)BB * HH, BB, HH, HH, 2);
  cvt_flat_kernel<<<dim3((EE * BB * HH + 255) / 256), 256, 0, stream>>>(fh, fh_bf, (long)EE * BB * HH);
  wmma_gemm_kernel<<<dim3((BB / 16) * (HH / 16) / 4, 1, EE), 128, 0, stream>>>(
      fh_bf, (long)BB * HH, few2, (long)HH * HH, F(P_FE_B2), HH, fe_out, (long)BB * HH, BB, HH, HH, 0);
  expert_sum_kernel<<<dim3((HH + 127) / 128, BB), 128, 0, stream>>>(fg, fe_out, fusion_sum, (long)BB * HH, HH);
  linear_kernel<<<dim3((174 + 127) / 128, BB), 128, 0, stream>>>(fusion_sum, F(P_AM_W1), F(P_AM_B1), ham, HH, 174, 1);
  linear_kernel<<<dim3(1, BB), 128, 0, stream>>>(ham, F(P_AM_W2), F(P_AM_B2), lm2, 174, 2, 0);
  scale01_kernel<<<dim3((HH + 127) / 128, BB), 128, 0, stream>>>(lm2, fusion_sum, fusion_final);

  // ---- prototype cost + sinkhorn OT + dissonance MLP ----
  proto_cost_kernel<<<dim3(BB), 160, 0, stream>>>(text_final, image_final, F(P_TP), F(P_IP), Cmat);
  sinkhorn_kernel<<<dim3(BB), 32, 0, stream>>>(Cmat, Tmat, 20, 0.05f);
  linear_kernel<<<dim3(1, BB), 128, 0, stream>>>(Tmat, F(P_DM_W1), F(P_DM_B1), dm_h, PP * PP, 128, 3);
  linear_kernel<<<dim3((HH + 127) / 128, BB), 128, 0, stream>>>(dm_h, F(P_DM_W2), F(P_DM_B2), diss, 128, HH, 0);

  // ---- final fuse ----
  concat3_kernel<<<dim3((640 + 127) / 128, BB), 128, 0, stream>>>(fusion_final, HH, diss, HH, nullptr, 0, final_cat);
  linear_kernel<<<dim3((HH + 127) / 128, BB), 128, 0, stream>>>(final_cat, F(P_FD_W1), F(P_FD_B1), fd_h, 640, HH, 1);
  linear_kernel<<<dim3((HH + 127) / 128, BB), 128, 0, stream>>>(fd_h, F(P_FD_W2), F(P_FD_B2), final_, HH, HH, 0);

  // ---- classifier heads ----
  linear_kernel<<<dim3((348 + 127) / 128, BB), 128, 0, stream>>>(text_final, F(P_CT_W1), F(P_CT_B1), ct_h, HH, 348, 1);
  linear_kernel<<<dim3(1, BB), 128, 0, stream>>>(ct_h, F(P_CT_W2), F(P_CT_B2), lt, 348, 1, 0);
  linear_kernel<<<dim3((348 + 127) / 128, BB), 128, 0, stream>>>(image_final, F(P_CI_W1), F(P_CI_B1), ci_h, HH, 348, 1);
  linear_kernel<<<dim3(1, BB), 128, 0, stream>>>(ci_h, F(P_CI_W2), F(P_CI_B2), li, 348, 1, 0);
  linear_kernel<<<dim3((348 + 127) / 128, BB), 128, 0, stream>>>(final_, F(P_CM_W1), F(P_CM_B1), cm_h, HH, 348, 1);
  linear_kernel<<<dim3(1, BB), 128, 0, stream>>>(cm_h, F(P_CM_W2), F(P_CM_B2), lm, 348, 1, 0);

  concat3_kernel<<<dim3(1, BB), 128, 0, stream>>>(lt, 1, li, 1, lm, 1, (float*)d_out);
}
